// LayerStacks_67645734912396
// MI455X (gfx1250) — compile-verified, matrix operand order checked by reading, not run on hardware
//
#include <hip/hip_runtime.h>

#define COUNT 8
#define L1    3072
#define L2    15
#define L3    32
#define NCOL  128          // COUNT*(L2+1), after folding l1f into l1_w
#define TM    64           // batch rows per workgroup (4 x 16-row WMMA tiles)
#define NRT   4            // row tiles per wave
#define KC    64           // K chunk (two K=32 WMMA steps)
#define LDSTR 72           // padded LDS row stride (bf16 elems): conflict-free b128 reads

typedef __attribute__((ext_vector_type(16))) __bf16         v16bf;
typedef __attribute__((ext_vector_type(8)))  float          v8f;
typedef __attribute__((ext_vector_type(16))) unsigned short v16u;
typedef __attribute__((ext_vector_type(8)))  unsigned short u16x8;
typedef __attribute__((ext_vector_type(4)))  unsigned short u16x4;

__device__ __forceinline__ unsigned short f2bf(float f) {
  unsigned int u = __float_as_uint(f);
  u += 0x7fffu + ((u >> 16) & 1u);          // round-to-nearest-even
  return (unsigned short)(u >> 16);
}
__device__ __forceinline__ float bf2f(unsigned short h) {
  return __uint_as_float(((unsigned int)h) << 16);
}
__device__ __forceinline__ float clamp01(float v) {
  return fminf(fmaxf(v, 0.0f), 1.0f);
}

// Build a 16-elem bf16 WMMA fragment: lane holds K {k0..k0+7} and {k0+16..k0+23}
// (k0 = chunk_k + (lane<16 ? 0 : 8)), as two 16-byte loads.
__device__ __forceinline__ v16bf frag16(const unsigned short* p) {
  u16x8 a = *(const u16x8*)(p);
  u16x8 b = *(const u16x8*)(p + 16);
  v16u v;
#pragma unroll
  for (int i = 0; i < 8; ++i) { v[i] = a[i]; v[i + 8] = b[i]; }
  return __builtin_bit_cast(v16bf, v);
}

// ---------------- prep: fuse l1f into l1_w, split fp32 -> bf16 hi/lo ----------------
__global__ void nnue_prep(const float* __restrict__ l1_w, const float* __restrict__ l1_b,
                          const float* __restrict__ l1f_w, const float* __restrict__ l1f_b,
                          unsigned short* __restrict__ whi, unsigned short* __restrict__ wlo,
                          float* __restrict__ bias) {
  const int i = blockIdx.x * 256 + threadIdx.x;
  if (i < NCOL * L1) {
    const int r = i / L1;
    const int k = i - r * L1;
    const float wv = l1_w[i] + l1f_w[(r & 15) * L1 + k];   // j = r % 16
    const unsigned short h = f2bf(wv);
    whi[i] = h;
    wlo[i] = f2bf(wv - bf2f(h));
  }
  if (i < NCOL) bias[i] = l1_b[i] + l1f_b[i & 15];
}

// ---------------- fused main kernel ----------------
__global__ void __launch_bounds__(256, 2)
nnue_main(const float* __restrict__ x,
          const int* __restrict__ lsi,
          const unsigned short* __restrict__ whi,
          const unsigned short* __restrict__ wlo,
          const float* __restrict__ bias,
          const float* __restrict__ l2_w,
          const float* __restrict__ l2_b,
          const float* __restrict__ out_w,
          const float* __restrict__ out_b,
          float* __restrict__ out) {
  __shared__ __align__(16) unsigned short s_xhi[2][TM][LDSTR];
  __shared__ __align__(16) unsigned short s_xlo[2][TM][LDSTR];
  __shared__ float s_l1s[TM][NCOL + 4];
  __shared__ float s_l2x[TM][L3];

  const int t    = threadIdx.x;
  const int wv   = t >> 5;        // wave id: owns output cols [16*wv, 16*wv+16) == bucket wv
  const int lane = t & 31;
  const int r0   = blockIdx.x * TM;

  const int nloc  = lane & 15;          // N (for B/C) or M (for A) within a tile
  const int ksel  = (lane >> 4) << 3;   // 0 or 8: K base select per half-wave
  const int rbase = (lane >> 4) << 3;   // C/D row base per half-wave

  const int ldr = t >> 4;               // cooperative-load row within a 16-row pass
  const int ldc = (t & 15) << 2;        // cooperative-load col (float), 0..60

  const unsigned short* bhi_row = whi + (size_t)(wv * 16 + nloc) * L1;
  const unsigned short* blo_row = wlo + (size_t)(wv * 16 + nloc) * L1;

  const float* xrow = x + (size_t)(r0 + ldr) * L1 + ldc;

  v8f acc[NRT];
#pragma unroll
  for (int rt = 0; rt < NRT; ++rt)
    acc[rt] = (v8f){0.f, 0.f, 0.f, 0.f, 0.f, 0.f, 0.f, 0.f};

  // ---- prologue: stage chunk 0 into buffer 0 ----
#pragma unroll
  for (int q = 0; q < NRT; ++q) {
    float4 xv = *(const float4*)(xrow + (size_t)q * 16 * L1);
    u16x4 h, l;
    float vv[4] = {xv.x, xv.y, xv.z, xv.w};
#pragma unroll
    for (int i = 0; i < 4; ++i) {
      const unsigned short hb = f2bf(vv[i]);
      h[i] = hb;
      l[i] = f2bf(vv[i] - bf2f(hb));
    }
    *(u16x4*)(&s_xhi[0][q * 16 + ldr][ldc]) = h;
    *(u16x4*)(&s_xlo[0][q * 16 + ldr][ldc]) = l;
  }
  __syncthreads();

  // ---- main loop: double-buffered, one barrier per chunk ----
  for (int kc = 0; kc < L1; kc += KC) {
    const int cur = (kc / KC) & 1;
    const bool have_next = (kc + KC) < L1;

    // issue next chunk's x loads before this chunk's WMMAs (explicit overlap)
    float4 xv[NRT];
    if (have_next) {
#pragma unroll
      for (int q = 0; q < NRT; ++q)
        xv[q] = *(const float4*)(xrow + (size_t)q * 16 * L1 + kc + KC);
      __builtin_prefetch(bhi_row + kc + KC + ksel, 0, 1);
      __builtin_prefetch(blo_row + kc + KC + ksel, 0, 1);
    }

#pragma unroll
    for (int kk = 0; kk < KC; kk += 32) {
      // B fragments loaded once per K-step, reused across 4 row tiles
      v16bf bhi = frag16(bhi_row + kc + kk + ksel);
      v16bf blo = frag16(blo_row + kc + kk + ksel);
#pragma unroll
      for (int rt = 0; rt < NRT; ++rt) {
        v16bf ahi = frag16(&s_xhi[cur][rt * 16 + nloc][0] + kk + ksel);
        v16bf alo = frag16(&s_xlo[cur][rt * 16 + nloc][0] + kk + ksel);
        // fp32 via bf16 hi/lo split: x*w ~= xh*wh + xh*wl + xl*wh
        acc[rt] = __builtin_amdgcn_wmma_f32_16x16x32_bf16(false, ahi, false, bhi, (short)0, acc[rt], false, false);
        acc[rt] = __builtin_amdgcn_wmma_f32_16x16x32_bf16(false, ahi, false, blo, (short)0, acc[rt], false, false);
        acc[rt] = __builtin_amdgcn_wmma_f32_16x16x32_bf16(false, alo, false, bhi, (short)0, acc[rt], false, false);
      }
    }

    // convert + store next chunk into the other buffer while cur is being read
    if (have_next) {
#pragma unroll
      for (int q = 0; q < NRT; ++q) {
        u16x4 h, l;
        float vv[4] = {xv[q].x, xv[q].y, xv[q].z, xv[q].w};
#pragma unroll
        for (int i = 0; i < 4; ++i) {
          const unsigned short hb = f2bf(vv[i]);
          h[i] = hb;
          l[i] = f2bf(vv[i] - bf2f(hb));
        }
        *(u16x4*)(&s_xhi[1 - cur][q * 16 + ldr][ldc]) = h;
        *(u16x4*)(&s_xlo[1 - cur][q * 16 + ldr][ldc]) = l;
      }
    }
    __syncthreads();
  }

  // spill l1s tiles (+fused bias) to LDS: C/D layout -> [row][col]
  {
    const int col  = wv * 16 + nloc;
    const float bc = bias[col];
#pragma unroll
    for (int rt = 0; rt < NRT; ++rt)
#pragma unroll
      for (int p = 0; p < 8; ++p)
        s_l1s[rt * 16 + rbase + p][col] = acc[rt][p] + bc;
  }
  __syncthreads();

  // layer 2: 4 threads per row, 8 of 32 outputs each (all 256 threads active)
  {
    const int i   = t >> 2;
    const int p   = t & 3;
    const int idx = lsi[r0 + i];
    const float* trow = &s_l1s[i][idx * 16];
    float l1x[2 * L2];
#pragma unroll
    for (int j = 0; j < L2; ++j) {
      const float v = trow[j];
      l1x[j]      = clamp01(v * v * (127.0f / 128.0f));
      l1x[L2 + j] = clamp01(v);
    }
#pragma unroll
    for (int q = 0; q < 8; ++q) {
      const int n = p * 8 + q;
      const float* wrow = l2_w + (size_t)(idx * L3 + n) * (2 * L2);
      float a = l2_b[idx * L3 + n];
#pragma unroll
      for (int j = 0; j < 2 * L2; ++j) a += l1x[j] * wrow[j];
      s_l2x[i][n] = clamp01(a);
    }
  }
  __syncthreads();

  // output layer + skip terms (l1c_out + l1f_out == fused col 15 of the bucket)
  if (t < TM) {
    const int idx = lsi[r0 + t];
    float s = out_b[idx];
#pragma unroll
    for (int n = 0; n < L3; ++n) s += s_l2x[t][n] * out_w[idx * L3 + n];
    out[r0 + t] = s + s_l1s[t][idx * 16 + L2];
  }
}

extern "C" void kernel_launch(void* const* d_in, const int* in_sizes, int n_in,
                              void* d_out, int out_size, void* d_ws, size_t ws_size,
                              hipStream_t stream) {
  const float* x     = (const float*)d_in[0];
  const int*   lsi   = (const int*)d_in[1];
  const float* l1_w  = (const float*)d_in[2];
  const float* l1_b  = (const float*)d_in[3];
  const float* l1f_w = (const float*)d_in[4];
  const float* l1f_b = (const float*)d_in[5];
  const float* l2_w  = (const float*)d_in[6];
  const float* l2_b  = (const float*)d_in[7];
  const float* out_w = (const float*)d_in[8];
  const float* out_b = (const float*)d_in[9];

  const int B = in_sizes[0] / L1;

  unsigned short* whi  = (unsigned short*)d_ws;
  unsigned short* wlo  = whi + (size_t)NCOL * L1;
  float*          bias = (float*)(wlo + (size_t)NCOL * L1);

  const int prepBlocks = (NCOL * L1 + 255) / 256;
  nnue_prep<<<prepBlocks, 256, 0, stream>>>(l1_w, l1_b, l1f_w, l1f_b, whi, wlo, bias);

  nnue_main<<<B / TM, 256, 0, stream>>>(x, lsi, whi, wlo, bias,
                                        l2_w, l2_b, out_w, out_b, (float*)d_out);
}
